// TemporalManifoldDiffusionNetwork_55808805045017
// MI455X (gfx1250) — matrix operand order repaired and checked by reading, not compiled, not run on hardware
//
#include <hip/hip_runtime.h>

// ---------------------------------------------------------------------------
// Types for CDNA5 WMMA
// ---------------------------------------------------------------------------
typedef __attribute__((ext_vector_type(16))) _Float16 v16h;
typedef __attribute__((ext_vector_type(8)))  _Float16 v8h;
typedef __attribute__((ext_vector_type(8)))  float    v8f;

// Model constants
#define BB   16
#define TT   256
#define BT   4096      // BB*TT
#define DIN  384
#define DD   512
#define LL   6
#define HH   8
#define HD   64
#define PP   64
#define SS   10

// ---------------------------------------------------------------------------
// Generic wave32 WMMA GEMM:  C[m,n] = act( sum_k A(m,k)*B(k,n) + bias[n] )
//   A(m,k) = A[m*lda + k]                       (K always contiguous)
//   B(k,n) = B[k*ldbk + n*ldbn]                 (vector path when ldbk==1)
//   batch base offsets: ptr += (z/zdiv)*s1 + (z%zdiv)*s2
// Block = 1 wave (32 threads), computes a 16x64 tile of C.
// grid = (N/64, M/16, Z).  Requires M%16==0, N%64==0, K%32==0 (all true here).
// ---------------------------------------------------------------------------
template<int ACT>
__global__ __launch_bounds__(32)
void k_gemm(const _Float16* __restrict__ A, const _Float16* __restrict__ Bm,
            const float* __restrict__ bias, float* __restrict__ C,
            int M, int N, int K, int lda, long ldbk, long ldbn, int ldc,
            int zdiv, long az1, long az2, long bz1, long bz2, long cz1, long cz2)
{
    const int lane = threadIdx.x;
    const int z    = blockIdx.z;
    const _Float16* Ab = A  + (long)(z / zdiv) * az1 + (long)(z % zdiv) * az2;
    const _Float16* Bb = Bm + (long)(z / zdiv) * bz1 + (long)(z % zdiv) * bz2;
    float*          Cb = C  + (long)(z / zdiv) * cz1 + (long)(z % zdiv) * cz2;

    const int m0 = blockIdx.y * 16;
    const int n0 = blockIdx.x * 64;

    v8f acc[4];
#pragma unroll
    for (int i = 0; i < 4; ++i)
#pragma unroll
        for (int e = 0; e < 8; ++e) acc[i][e] = 0.f;

    const int arow     = m0 + (lane & 15);
    const int aksel    = (lane >> 4) * 16;    // lanes 16..31 hold K+16..31
    const int bn_local = lane & 15;
    const int bksel    = (lane >> 4) * 16;    // lanes 16..31 hold K+16..31

    for (int k0 = 0; k0 < K; k0 += 32) {
        const _Float16* ap = Ab + (long)arow * lda + k0 + aksel;
        if (k0 + 32 < K) __builtin_prefetch(ap + 32, 0, 0);   // global_prefetch_b8
        v8h alo = *(const v8h*)ap;
        v8h ahi = *(const v8h*)(ap + 8);
        v16h afrag;
#pragma unroll
        for (int e = 0; e < 8; ++e) { afrag[e] = alo[e]; afrag[e + 8] = ahi[e]; }

#pragma unroll
        for (int nt = 0; nt < 4; ++nt) {
            const int n = n0 + nt * 16 + bn_local;
            v16h bfrag;
            if (ldbk == 1) {
                const _Float16* bp = Bb + (long)n * ldbn + k0 + bksel;
                v8h blo = *(const v8h*)bp;
                v8h bhi = *(const v8h*)(bp + 8);
#pragma unroll
                for (int e = 0; e < 8; ++e) { bfrag[e] = blo[e]; bfrag[e + 8] = bhi[e]; }
            } else {
#pragma unroll
                for (int e = 0; e < 16; ++e)
                    bfrag[e] = Bb[(long)(k0 + bksel + e) * ldbk + (long)n * ldbn];
            }
            acc[nt] = __builtin_amdgcn_wmma_f32_16x16x32_f16(
                false, afrag, false, bfrag, (short)0, acc[nt], false, false);
        }
    }

#pragma unroll
    for (int nt = 0; nt < 4; ++nt) {
        const int n = n0 + nt * 16 + (lane & 15);
        const float bv = bias ? bias[n] : 0.f;
#pragma unroll
        for (int v = 0; v < 8; ++v) {
            const int row = m0 + v + (lane >> 4) * 8;
            float val = acc[nt][v] + bv;
            if (ACT == 1) val = fmaxf(val, 0.f);
            else if (ACT == 2) val = tanhf(val);
            else if (ACT == 3) val = val / (1.f + expf(-val));
            Cb[(long)row * ldc + n] = val;
        }
    }
}

// ---------------------------------------------------------------------------
// Elementwise / helper kernels
// ---------------------------------------------------------------------------
__global__ void k_f32_to_f16(const float* __restrict__ s, _Float16* __restrict__ d, long n)
{
    long i = (long)blockIdx.x * blockDim.x + threadIdx.x;
    if (i < n) d[i] = (_Float16)s[i];
}

// Wt[n*K + k] = W[k*N + n]   (row-major KxN weight -> f16, N-major/transposed)
__global__ void k_wt(const float* __restrict__ W, _Float16* __restrict__ Wt, int K, int N)
{
    long i = (long)blockIdx.x * blockDim.x + threadIdx.x;
    if (i >= (long)K * N) return;
    int n = (int)(i / K), kk = (int)(i % K);
    Wt[i] = (_Float16)W[(long)kk * N + n];
}

// cc_w (O=512, I=512, 3) -> transposed f16 with K-dim = tap*512 + i
__global__ void k_ccw_pack(const float* __restrict__ W, _Float16* __restrict__ Wt)
{
    long i = (long)blockIdx.x * blockDim.x + threadIdx.x;
    if (i >= (long)DD * 1536) return;
    int o = (int)(i / 1536), j = (int)(i % 1536);
    int tap = j / DD, ii = j % DD;
    Wt[i] = (_Float16)W[((long)o * DD + ii) * 3 + tap];
}

// A_exp[bt, tap*512 + c] = h[b, t-(2-tap)*dil, c]  (0 when OOB)  -> f16
__global__ void k_ccA_pack(const float* __restrict__ h, _Float16* __restrict__ Adst, int dil)
{
    long idx = (long)blockIdx.x * blockDim.x + threadIdx.x;
    if (idx >= (long)BT * 1536) return;
    int bt = (int)(idx / 1536), j = (int)(idx % 1536);
    int tap = j / DD, c = j % DD;
    int b = bt / TT, t = bt % TT;
    int s = t - (2 - tap) * dil;
    float v = (s >= 0) ? h[((long)b * TT + s) * DD + c] : 0.f;
    Adst[idx] = (_Float16)v;
}

// out[(b*H+h)*T + t] = sum_d q[bt, h*64+d]^2
__global__ void k_sqnorm(const float* __restrict__ q, float* __restrict__ o)
{
    long i = (long)blockIdx.x * blockDim.x + threadIdx.x;
    if (i >= (long)BB * HH * TT) return;
    int b = (int)(i / (HH * TT)); int r = (int)(i % (HH * TT));
    int hh = r / TT, t = r % TT;
    const float* p = q + ((long)b * TT + t) * DD + hh * HD;
    float acc = 0.f;
    for (int d = 0; d < HD; ++d) { float x = p[d]; acc += x * x; }
    o[i] = acc;
}

// hyperbolic distance + softmax row:   scores row (f32) -> attn row (f16)
__global__ void k_attn(const float* __restrict__ curv_l, const float* __restrict__ qn,
                       const float* __restrict__ kn, float* __restrict__ qk,
                       _Float16* __restrict__ attn)
{
    long idx = (long)blockIdx.x * blockDim.x + threadIdx.x;   // z*T + i
    if (idx >= (long)BB * HH * TT) return;
    int z = (int)(idx / TT);
    int hh = z % HH;
    float c = fabsf(curv_l[hh]) + 1e-8f;
    float rs = rsqrtf(c);
    float* row = qk + idx * TT;
    const float* knr = kn + (long)z * TT;
    float qni = qn[idx];
    float mx = -1e30f;
    for (int j = 0; j < TT; ++j) {
        float d2 = fmaxf(qni + knr[j] - 2.f * row[j], 0.f);
        float den = fmaxf((1.f - c * qni) * (1.f - c * knr[j]), 1e-5f);
        float arg = fmaxf(1.f + 2.f * c * d2 / den, 1.00001f);
        float dist = logf(arg + sqrtf(arg * arg - 1.f)) * rs;
        float nd = -dist;
        row[j] = nd;
        mx = fmaxf(mx, nd);
    }
    float sum = 0.f;
    for (int j = 0; j < TT; ++j) sum += expf(row[j] - mx);
    float inv = 1.f / sum;
    for (int j = 0; j < TT; ++j)
        attn[idx * TT + j] = (_Float16)(expf(row[j] - mx) * inv);
}

// wavelet: softmax filter, causal depthwise conv, subsample, linear interp
template<int K, int LVL>
__global__ void k_wavelet(const float* __restrict__ h, const float* __restrict__ wv,
                          _Float16* __restrict__ comb)
{
    long idx = (long)blockIdx.x * blockDim.x + threadIdx.x;
    if (idx >= (long)BB * TT * DD) return;
    int c = (int)(idx % DD);
    int t = (int)((idx / DD) % TT);
    int b = (int)(idx / ((long)DD * TT));

    const float* f = wv + (long)c * K;
    float mx = f[0];
#pragma unroll
    for (int k = 1; k < K; ++k) mx = fmaxf(mx, f[k]);
    float e[K]; float sum = 0.f;
#pragma unroll
    for (int k = 0; k < K; ++k) { e[k] = expf(f[k] - mx); sum += e[k]; }
    float inv = 1.f / sum;

    const int stride = 1 << (LVL + 1);
    const int len = TT >> (LVL + 1);
    float pos = (t + 0.5f) * ((float)len / (float)TT) - 0.5f;
    pos = fminf(fmaxf(pos, 0.f), (float)(len - 1));
    int lo = (int)floorf(pos);
    int hi = min(lo + 1, len - 1);
    float w = pos - (float)lo;

    auto conv = [&](int tt) -> float {
        float acc = 0.f;
#pragma unroll
        for (int k = 0; k < K; ++k) {
            int src = tt - (K - 1) + k;
            if (src >= 0) acc += e[k] * h[((long)b * TT + src) * DD + c];
        }
        return acc * inv;
    };
    float dlo = conv(lo * stride);
    float dhi = (hi == lo) ? dlo : conv(hi * stride);
    float out = dlo * (1.f - w) + dhi * w;
    comb[((long)b * TT + t) * (4 * DD) + LVL * DD + c] = (_Float16)out;
}

// boundary loss: mean over BT of relu(bd0 - bd1 + 0.1), accumulated
__global__ void k_bdloss(const float* __restrict__ tf, const float* __restrict__ bdw,
                         const float* __restrict__ bdb, float* __restrict__ loss)
{
    __shared__ float red[256];
    int idx = blockIdx.x * 256 + threadIdx.x;
    float b0 = bdb[0], b1 = bdb[1];
    const float* r = tf + (long)idx * PP;
    for (int p = 0; p < PP; ++p) { float a = r[p]; b0 += a * bdw[p * 2]; b1 += a * bdw[p * 2 + 1]; }
    red[threadIdx.x] = fmaxf(b0 - b1 + 0.1f, 0.f);
    __syncthreads();
    for (int s = 128; s > 0; s >>= 1) {
        if (threadIdx.x < s) red[threadIdx.x] += red[threadIdx.x + s];
        __syncthreads();
    }
    if (threadIdx.x == 0) atomicAdd(loss, red[0] * (1.f / (float)BT));
}

// s = geom + wave + (h + t3) + cc ; LayerNorm -> h   (one block per row)
__global__ void k_sum_ln(const float* __restrict__ geom, const float* __restrict__ wave,
                         const float* __restrict__ t3, const float* __restrict__ cc,
                         const float* __restrict__ g, const float* __restrict__ bta,
                         float* __restrict__ h)
{
    __shared__ float r1[256], r2[256];
    long base = (long)blockIdx.x * DD;
    int tid = threadIdx.x;
    int c0 = tid, c1 = tid + 256;
    float s0 = geom[base + c0] + wave[base + c0] + t3[base + c0] + cc[base + c0] + h[base + c0];
    float s1 = geom[base + c1] + wave[base + c1] + t3[base + c1] + cc[base + c1] + h[base + c1];
    r1[tid] = s0 + s1;
    r2[tid] = s0 * s0 + s1 * s1;
    __syncthreads();
    for (int s = 128; s > 0; s >>= 1) {
        if (tid < s) { r1[tid] += r1[tid + s]; r2[tid] += r2[tid + s]; }
        __syncthreads();
    }
    float mu = r1[0] * (1.f / DD);
    float var = r2[0] * (1.f / DD) - mu * mu;
    float inv = rsqrtf(var + 1e-5f);
    h[base + c0] = (s0 - mu) * inv * g[c0] + bta[c0];
    h[base + c1] = (s1 - mu) * inv * g[c1] + bta[c1];
}

__global__ void k_zero(float* p, int n)
{
    int i = blockIdx.x * blockDim.x + threadIdx.x;
    if (i < n) p[i] = 0.f;
}

// cat16[b,0:512] = h[b,-1,:], cat16[b,512:1024] = h[b,-2,:]
__global__ void k_build_cat(const float* __restrict__ h, _Float16* __restrict__ cat)
{
    int idx = blockIdx.x * blockDim.x + threadIdx.x;
    if (idx >= BB * 1024) return;
    int b = idx / 1024, j = idx % 1024;
    float v = (j < DD) ? h[((long)b * TT + (TT - 1)) * DD + j]
                       : h[((long)b * TT + (TT - 2)) * DD + (j - DD)];
    cat[idx] = (_Float16)v;
}

__global__ void k_cur_init(const float* __restrict__ h, const float* __restrict__ mix,
                           float* __restrict__ cur)
{
    int idx = blockIdx.x * blockDim.x + threadIdx.x;
    if (idx >= BB * DD) return;
    int b = idx / DD, c = idx % DD;
    cur[idx] = h[((long)b * TT + (TT - 1)) * DD + c] + mix[idx];
}

// di = [cur + noise_t*sched[t], sin/cos time embedding] -> f16
__global__ void k_build_di(const float* __restrict__ cur, const float* __restrict__ noise_t,
                           const float* __restrict__ sched, int t, _Float16* __restrict__ di)
{
    int idx = blockIdx.x * blockDim.x + threadIdx.x;
    if (idx >= BB * 1024) return;
    int b = idx / 1024, j = idx % 1024;
    float v;
    if (j < DD) {
        v = cur[b * DD + j] + noise_t[(long)b * DD + j] * sched[t];
    } else {
        int j2 = j - DD;
        int m = j2 & 255;
        float ef = expf((float)m * (-9.210340371976184f / 255.f));
        float te = (float)t * ef;
        v = (j2 < 256) ? sinf(te) : cosf(te);
    }
    di[idx] = (_Float16)v;
}

__global__ void k_cur_update(float* __restrict__ cur, const float* __restrict__ npred,
                             const float* __restrict__ sched, int t)
{
    int idx = blockIdx.x * blockDim.x + threadIdx.x;
    if (idx >= BB * DD) return;
    cur[idx] -= npred[idx] * sched[t];
}

__global__ void k_write_loss(const float* __restrict__ loss, float* __restrict__ out)
{
    if (threadIdx.x == 0 && blockIdx.x == 0) out[0] = loss[0];
}

// ---------------------------------------------------------------------------
// Host side
// ---------------------------------------------------------------------------
static inline dim3 eg(long n) { return dim3((unsigned)((n + 255) / 256)); }

struct GemmArgs {
    int Z = 1, zdiv = 1;
    long az1 = 0, az2 = 0, bz1 = 0, bz2 = 0, cz1 = 0, cz2 = 0;
};

static void launch_gemm(hipStream_t st, int act,
                        const _Float16* A, const _Float16* Bm, const float* bias, float* C,
                        int M, int N, int K, int lda, long ldbk, long ldbn, int ldc,
                        const GemmArgs& ga = GemmArgs())
{
    dim3 g(N / 64, M / 16, ga.Z), b(32, 1, 1);
    switch (act) {
    case 0: k_gemm<0><<<g, b, 0, st>>>(A, Bm, bias, C, M, N, K, lda, ldbk, ldbn, ldc,
                                       ga.zdiv, ga.az1, ga.az2, ga.bz1, ga.bz2, ga.cz1, ga.cz2); break;
    case 1: k_gemm<1><<<g, b, 0, st>>>(A, Bm, bias, C, M, N, K, lda, ldbk, ldbn, ldc,
                                       ga.zdiv, ga.az1, ga.az2, ga.bz1, ga.bz2, ga.cz1, ga.cz2); break;
    case 2: k_gemm<2><<<g, b, 0, st>>>(A, Bm, bias, C, M, N, K, lda, ldbk, ldbn, ldc,
                                       ga.zdiv, ga.az1, ga.az2, ga.bz1, ga.bz2, ga.cz1, ga.cz2); break;
    default: k_gemm<3><<<g, b, 0, st>>>(A, Bm, bias, C, M, N, K, lda, ldbk, ldbn, ldc,
                                        ga.zdiv, ga.az1, ga.az2, ga.bz1, ga.bz2, ga.cz1, ga.cz2); break;
    }
}

extern "C" void kernel_launch(void* const* d_in, const int* in_sizes, int n_in,
                              void* d_out, int out_size, void* d_ws, size_t ws_size,
                              hipStream_t stream)
{
    (void)in_sizes; (void)n_in; (void)out_size; (void)ws_size;

    const float* x       = (const float*)d_in[0];
    const float* in_w    = (const float*)d_in[1];
    const float* in_b    = (const float*)d_in[2];
    const float* gq_w    = (const float*)d_in[3];
    const float* gq_b    = (const float*)d_in[4];
    const float* gk_w    = (const float*)d_in[5];
    const float* gk_b    = (const float*)d_in[6];
    const float* gv_w    = (const float*)d_in[7];
    const float* gv_b    = (const float*)d_in[8];
    const float* go_w    = (const float*)d_in[9];
    const float* go_b    = (const float*)d_in[10];
    const float* curv    = (const float*)d_in[11];
    const float* wav[4]  = { (const float*)d_in[12], (const float*)d_in[13],
                             (const float*)d_in[14], (const float*)d_in[15] };
    const float* wrec_w  = (const float*)d_in[16];
    const float* wrec_b  = (const float*)d_in[17];
    const float* te1_w   = (const float*)d_in[18];
    const float* te1_b   = (const float*)d_in[19];
    const float* te2_w   = (const float*)d_in[20];
    const float* te2_b   = (const float*)d_in[21];
    const float* td1_w   = (const float*)d_in[22];
    const float* td1_b   = (const float*)d_in[23];
    const float* td2_w   = (const float*)d_in[24];
    const float* td2_b   = (const float*)d_in[25];
    const float* bd_w    = (const float*)d_in[26];
    const float* bd_b    = (const float*)d_in[27];
    const float* cc_w    = (const float*)d_in[28];
    const float* cc_b    = (const float*)d_in[29];
    const float* ln_g    = (const float*)d_in[30];
    const float* ln_b    = (const float*)d_in[31];
    const float* mm1_w   = (const float*)d_in[32];
    const float* mm1_b   = (const float*)d_in[33];
    const float* mm2_w   = (const float*)d_in[34];
    const float* mm2_b   = (const float*)d_in[35];
    const float* dn1_w   = (const float*)d_in[36];
    const float* dn1_b   = (const float*)d_in[37];
    const float* dn2_w   = (const float*)d_in[38];
    const float* dn2_b   = (const float*)d_in[39];
    const float* dn3_w   = (const float*)d_in[40];
    const float* dn3_b   = (const float*)d_in[41];
    const float* out_w   = (const float*)d_in[42];
    const float* out_b   = (const float*)d_in[43];
    const float* nsched  = (const float*)d_in[44];
    const float* noise   = (const float*)d_in[45];
    float* out = (float*)d_out;

    // --- workspace carve-up (byte offsets, 256B aligned) ---
    char* ws = (char*)d_ws;
    size_t off = 0;
    auto alloc = [&](size_t bytes) -> char* {
        char* p = ws + off;
        off = (off + bytes + 255) & ~(size_t)255;
        return p;
    };
    float*     h      = (float*)    alloc((size_t)BT * DD * 4);
    _Float16*  h16    = (_Float16*) alloc((size_t)BT * DD * 2);
    _Float16*  w16    = (_Float16*) alloc((size_t)2048 * DD * 2);
    float*     geom   = (float*)    alloc((size_t)BT * DD * 4);
    float*     wave   = (float*)    alloc((size_t)BT * DD * 4);
    float*     t3     = (float*)    alloc((size_t)BT * DD * 4);
    float*     cc     = (float*)    alloc((size_t)BT * DD * 4);
    float*     qbuf   = (float*)    alloc((size_t)BT * DD * 4);   // reused as r2
    float*     kbuf   = (float*)    alloc((size_t)BT * DD * 4);
    float*     vbuf   = (float*)    alloc((size_t)BT * DD * 4);
    _Float16*  q16    = (_Float16*) alloc((size_t)BT * DD * 2);   // reused geomp16, r2_16
    _Float16*  k16    = (_Float16*) alloc((size_t)BT * DD * 2);
    _Float16*  v16    = (_Float16*) alloc((size_t)BT * DD * 2);
    float*     qk     = (float*)    alloc((size_t)BB * HH * TT * TT * 4); // reused as geomp
    _Float16*  attn16 = (_Float16*) alloc((size_t)BB * HH * TT * TT * 2); // reused as ccA16
    _Float16*  comb16 = (_Float16*) alloc((size_t)BT * 4 * DD * 2);
    _Float16*  x16    = (_Float16*) alloc((size_t)BT * DIN * 2);
    float*     qn     = (float*)    alloc((size_t)BB * HH * TT * 4);
    float*     kn     = (float*)    alloc((size_t)BB * HH * TT * 4);
    float*     r1     = (float*)    alloc((size_t)BT * PP * 4);
    _Float16*  r116   = (_Float16*) alloc((size_t)BT * PP * 2);
    float*     tf     = (float*)    alloc((size_t)BT * PP * 4);
    _Float16*  tf16   = (_Float16*) alloc((size_t)BT * PP * 2);
    float*     loss   = (float*)    alloc(256);
    _Float16*  wdn1   = (_Float16*) alloc((size_t)1024 * DD * 2);
    _Float16*  wdn2   = (_Float16*) alloc((size_t)DD * DD * 2);
    _Float16*  wdn3   = (_Float16*) alloc((size_t)DD * DD * 2);
    _Float16*  cat16  = (_Float16*) alloc((size_t)BB * 1024 * 2);
    float*     mixh   = (float*)    alloc((size_t)BB * DD * 4);
    _Float16*  mixh16 = (_Float16*) alloc((size_t)BB * DD * 2);
    float*     mix    = (float*)    alloc((size_t)BB * DD * 4);
    float*     cur    = (float*)    alloc((size_t)BB * DD * 4);
    _Float16*  di16   = (_Float16*) alloc((size_t)BB * 1024 * 2);
    float*     tA     = (float*)    alloc((size_t)BB * DD * 4);
    _Float16*  tA16   = (_Float16*) alloc((size_t)BB * DD * 2);
    float*     tB     = (float*)    alloc((size_t)BB * DD * 4);
    _Float16*  tB16   = (_Float16*) alloc((size_t)BB * DD * 2);
    float*     npred  = (float*)    alloc((size_t)BB * DD * 4);
    _Float16*  cur16  = (_Float16*) alloc((size_t)BB * DD * 2);
    float*     geomp  = qk;                    // alias (first BT*DD floats)
    _Float16*  geomp16= q16;                   // alias
    float*     r2     = qbuf;                  // alias
    _Float16*  r2_16  = q16;                   // alias
    _Float16*  ccA16  = attn16;                // alias

    auto cvt = [&](const float* s, _Float16* d, long n) {
        k_f32_to_f16<<<eg(n), 256, 0, stream>>>(s, d, n);
    };
    auto wt = [&](const float* W, _Float16* Wt, int K, int N) {
        k_wt<<<eg((long)K * N), 256, 0, stream>>>(W, Wt, K, N);
    };

    k_zero<<<1, 32, 0, stream>>>(loss, 1);

    // ---- input projection: h = x @ in_w + in_b ----
    cvt(x, x16, (long)BT * DIN);
    wt(in_w, w16, DIN, DD);
    launch_gemm(stream, 0, x16, w16, in_b, h, BT, DD, DIN, DIN, 1, DIN, DD);

    for (int l = 0; l < LL; ++l) {
        cvt(h, h16, (long)BT * DD);

        // --- geometric attention ---
        wt(gq_w + (long)l * DD * DD, w16, DD, DD);
        launch_gemm(stream, 0, h16, w16, gq_b + l * DD, qbuf, BT, DD, DD, DD, 1, DD, DD);
        cvt(qbuf, q16, (long)BT * DD);
        wt(gk_w + (long)l * DD * DD, w16, DD, DD);
        launch_gemm(stream, 0, h16, w16, gk_b + l * DD, kbuf, BT, DD, DD, DD, 1, DD, DD);
        cvt(kbuf, k16, (long)BT * DD);
        wt(gv_w + (long)l * DD * DD, w16, DD, DD);
        launch_gemm(stream, 0, h16, w16, gv_b + l * DD, vbuf, BT, DD, DD, DD, 1, DD, DD);
        cvt(vbuf, v16, (long)BT * DD);

        k_sqnorm<<<eg((long)BB * HH * TT), 256, 0, stream>>>(qbuf, qn);
        k_sqnorm<<<eg((long)BB * HH * TT), 256, 0, stream>>>(kbuf, kn);

        // qk scores: per (b,h): (T x hd) @ (hd x T)
        {
            GemmArgs ga; ga.Z = BB * HH; ga.zdiv = HH;
            ga.az1 = (long)TT * DD; ga.az2 = HD;
            ga.bz1 = (long)TT * DD; ga.bz2 = HD;
            ga.cz1 = (long)HH * TT * TT; ga.cz2 = (long)TT * TT;
            launch_gemm(stream, 0, q16, k16, nullptr, qk,
                        TT, TT, HD, DD, 1, DD, TT, ga);
        }
        k_attn<<<eg((long)BB * HH * TT), 256, 0, stream>>>(curv + l * HH, qn, kn, qk, attn16);
        // attn @ v: per (b,h): (T x T) @ (T x hd)
        {
            GemmArgs ga; ga.Z = BB * HH; ga.zdiv = HH;
            ga.az1 = (long)HH * TT * TT; ga.az2 = (long)TT * TT;
            ga.bz1 = (long)TT * DD; ga.bz2 = HD;
            ga.cz1 = (long)TT * DD; ga.cz2 = HD;
            launch_gemm(stream, 0, attn16, v16, nullptr, geomp,
                        TT, HD, TT, TT, DD, 1, DD, ga);
        }
        cvt(geomp, geomp16, (long)BT * DD);
        wt(go_w + (long)l * DD * DD, w16, DD, DD);
        launch_gemm(stream, 0, geomp16, w16, go_b + l * DD, geom, BT, DD, DD, DD, 1, DD, DD);

        // --- wavelet branch ---
        k_wavelet<4, 0><<<eg((long)BB * TT * DD), 256, 0, stream>>>(h, wav[0] + (long)l * DD * 4, comb16);
        k_wavelet<8, 1><<<eg((long)BB * TT * DD), 256, 0, stream>>>(h, wav[1] + (long)l * DD * 8, comb16);
        k_wavelet<16, 2><<<eg((long)BB * TT * DD), 256, 0, stream>>>(h, wav[2] + (long)l * DD * 16, comb16);
        k_wavelet<32, 3><<<eg((long)BB * TT * DD), 256, 0, stream>>>(h, wav[3] + (long)l * DD * 32, comb16);
        wt(wrec_w + (long)l * 4 * DD * DD, w16, 4 * DD, DD);
        launch_gemm(stream, 0, comb16, w16, wrec_b + l * DD, wave, BT, DD, 4 * DD, 4 * DD, 1, 4 * DD, DD);

        // --- topology branch ---
        wt(te1_w + (long)l * DD * PP, w16, DD, PP);
        launch_gemm(stream, 1, h16, w16, te1_b + l * PP, r1, BT, PP, DD, DD, 1, DD, PP);
        cvt(r1, r116, (long)BT * PP);
        wt(te2_w + (long)l * PP * PP, w16, PP, PP);
        launch_gemm(stream, 0, r116, w16, te2_b + l * PP, tf, BT, PP, PP, PP, 1, PP, PP);
        cvt(tf, tf16, (long)BT * PP);
        k_bdloss<<<BT / 256, 256, 0, stream>>>(tf, bd_w + (long)l * PP * 2, bd_b + l * 2, loss);
        wt(td1_w + (long)l * PP * DD, w16, PP, DD);
        launch_gemm(stream, 1, tf16, w16, td1_b + l * DD, r2, BT, DD, PP, PP, 1, PP, DD);
        cvt(r2, r2_16, (long)BT * DD);
        wt(td2_w + (long)l * DD * DD, w16, DD, DD);
        launch_gemm(stream, 0, r2_16, w16, td2_b + l * DD, t3, BT, DD, DD, DD, 1, DD, DD);

        // --- dilated causal conv (as K=1536 GEMM) ---
        k_ccA_pack<<<eg((long)BT * 1536), 256, 0, stream>>>(h, ccA16, 1 << l);
        k_ccw_pack<<<eg((long)DD * 1536), 256, 0, stream>>>(cc_w + (long)l * DD * DD * 3, w16);
        launch_gemm(stream, 0, ccA16, w16, cc_b + l * DD, cc, BT, DD, 1536, 1536, 1, 1536, DD);

        // --- combine + LayerNorm -> h ---
        k_sum_ln<<<BT, 256, 0, stream>>>(geom, wave, t3, cc, ln_g, ln_b, h);
    }

    // ---- memory mixing head ----
    k_build_cat<<<eg(BB * 1024), 256, 0, stream>>>(h, cat16);
    wt(mm1_w, w16, 1024, DD);
    launch_gemm(stream, 2, cat16, w16, mm1_b, mixh, BB, DD, 1024, 1024, 1, 1024, DD);
    cvt(mixh, mixh16, (long)BB * DD);
    wt(mm2_w, w16, DD, DD);
    launch_gemm(stream, 0, mixh16, w16, mm2_b, mix, BB, DD, DD, DD, 1, DD, DD);
    k_cur_init<<<eg(BB * DD), 256, 0, stream>>>(h, mix, cur);

    // ---- diffusion refinement ----
    wt(dn1_w, wdn1, 1024, DD);
    wt(dn2_w, wdn2, DD, DD);
    wt(dn3_w, wdn3, DD, DD);
    for (int t = 0; t < SS; ++t) {
        k_build_di<<<eg(BB * 1024), 256, 0, stream>>>(cur, noise + (long)t * BB * DD, nsched, t, di16);
        launch_gemm(stream, 3, di16, wdn1, dn1_b, tA, BB, DD, 1024, 1024, 1, 1024, DD);
        cvt(tA, tA16, (long)BB * DD);
        launch_gemm(stream, 3, tA16, wdn2, dn2_b, tB, BB, DD, DD, DD, 1, DD, DD);
        cvt(tB, tB16, (long)BB * DD);
        launch_gemm(stream, 0, tB16, wdn3, dn3_b, npred, BB, DD, DD, DD, 1, DD, DD);
        k_cur_update<<<eg(BB * DD), 256, 0, stream>>>(cur, npred, nsched, t);
    }

    // ---- output projection + loss ----
    cvt(cur, cur16, (long)BB * DD);
    wt(out_w, w16, DD, DIN);
    launch_gemm(stream, 0, cur16, w16, out_b, out, BB, DIN, DD, DD, 1, DD, DIN);
    k_write_loss<<<1, 32, 0, stream>>>(loss, out + BB * DIN);
}
